// SelfAttention_39582418600136
// MI455X (gfx1250) — matrix operand order repaired and checked by reading, not compiled, not run on hardware
//
#include <hip/hip_runtime.h>

#define BATCH 8
#define CCH   256
#define CQKD  32
#define NPIX  4096

typedef __attribute__((ext_vector_type(16))) __bf16 v16bf;
typedef __attribute__((ext_vector_type(8)))  float  v8f;
typedef __attribute__((ext_vector_type(4)))  unsigned int u32x4;

union BV {
  v16bf v;
  u32x4 u[2];
  unsigned short s[16];
};

static __device__ __forceinline__ unsigned short f2bf(float f) {
  unsigned u = __float_as_uint(f);
  unsigned r = u + 0x7FFFu + ((u >> 16) & 1u);   // round-to-nearest-even
  return (unsigned short)(r >> 16);
}

static __device__ __forceinline__ v8f wmma_bf16(v16bf a, v16bf b, v8f c) {
  return __builtin_amdgcn_wmma_f32_16x16x32_bf16(false, a, false, b, (short)0, c,
                                                 false, false);
}

// LDS byte offset of a __shared__ object: flat LDS addresses carry the
// wave-relative LDS offset in addr[31:0] (CDNA5 aperture mapping).
static __device__ __forceinline__ unsigned lds_off(const void* p) {
  return (unsigned)(uintptr_t)p;
}

// CDNA5 async DMA: global -> LDS, 16B per lane, tracked by ASYNCcnt.
static __device__ __forceinline__ void async_ld_b128(unsigned lo, const void* g) {
  asm volatile("global_load_async_to_lds_b128 %0, %1, off"
               :: "v"(lo), "v"(g) : "memory");
}
#define WAIT_ASYNC(n) asm volatile("s_wait_asynccnt %0" :: "n"(n) : "memory")
#define MEMBAR()      asm volatile("" ::: "memory")

// ---------------- kernel 0: convert weights to bf16 ----------------
__global__ void k_wcvt(const float* __restrict__ Wq, const float* __restrict__ Wk,
                       const float* __restrict__ Wv,
                       unsigned short* __restrict__ wq16,
                       unsigned short* __restrict__ wk16,
                       unsigned short* __restrict__ wv16) {
  int i = blockIdx.x * blockDim.x + threadIdx.x;
  if (i < 8192)        wq16[i]         = f2bf(Wq[i]);
  else if (i < 16384)  wk16[i - 8192]  = f2bf(Wk[i - 8192]);
  else if (i < 81920)  wv16[i - 16384] = f2bf(Wv[i - 16384]);
}

// ---------------- kernel 1: x[B][C][N] f32 -> Xp[B][N][C] bf16 ----------------
__global__ void k_xpose(const float* __restrict__ x, unsigned short* __restrict__ xp) {
  __shared__ unsigned short t[32][33];
  int b  = blockIdx.z;
  int c0 = blockIdx.y * 32;
  int n0 = blockIdx.x * 32;
  int tx = threadIdx.x, ty = threadIdx.y;
#pragma unroll
  for (int k = 0; k < 4; ++k) {
    int c = c0 + ty + k * 8;
    t[ty + k * 8][tx] = f2bf(x[((size_t)b * CCH + c) * NPIX + n0 + tx]);
  }
  __syncthreads();
#pragma unroll
  for (int k = 0; k < 4; ++k) {
    int n = n0 + ty + k * 8;
    xp[((size_t)b * NPIX + n) * CCH + c0 + tx] = t[tx][ty + k * 8];
  }
}

// ---------------- kernel 2: Q,K projections (WMMA) ----------------
__global__ void __launch_bounds__(32) k_proj_qk(
    const unsigned short* __restrict__ xp,
    const unsigned short* __restrict__ wq, const unsigned short* __restrict__ wk,
    const float* __restrict__ bq, const float* __restrict__ bk,
    unsigned short* __restrict__ Qb, unsigned short* __restrict__ Kb) {
  int l = threadIdx.x, half = l >> 4, lr = l & 15;
  int b  = blockIdx.x >> 8;
  int m0 = (blockIdx.x & 255) * 16;

  v8f zero = {};
  v8f aq0 = zero, aq1 = zero, ak0 = zero, ak1 = zero;
  const u32x4* arow = (const u32x4*)(xp + ((size_t)(b * NPIX + m0 + lr)) * CCH);
  const u32x4* wq0p = (const u32x4*)(wq + (size_t)lr * CCH);
  const u32x4* wq1p = (const u32x4*)(wq + (size_t)(16 + lr) * CCH);
  const u32x4* wk0p = (const u32x4*)(wk + (size_t)lr * CCH);
  const u32x4* wk1p = (const u32x4*)(wk + (size_t)(16 + lr) * CCH);

#pragma unroll
  for (int kc = 0; kc < 8; ++kc) {
    int u = kc * 4;
    BV A, B0, B1, B2, B3;
    A.u[0]  = arow[u + half * 2];  A.u[1]  = arow[u + half * 2 + 1];
    B0.u[0] = wq0p[u + half * 2];  B0.u[1] = wq0p[u + half * 2 + 1];
    B1.u[0] = wq1p[u + half * 2];  B1.u[1] = wq1p[u + half * 2 + 1];
    B2.u[0] = wk0p[u + half * 2];  B2.u[1] = wk0p[u + half * 2 + 1];
    B3.u[0] = wk1p[u + half * 2];  B3.u[1] = wk1p[u + half * 2 + 1];
    aq0 = wmma_bf16(A.v, B0.v, aq0);
    aq1 = wmma_bf16(A.v, B1.v, aq1);
    ak0 = wmma_bf16(A.v, B2.v, ak0);
    ak1 = wmma_bf16(A.v, B3.v, ak1);
  }
  float bql = bq[lr], bqh = bq[16 + lr], bkl = bk[lr], bkh = bk[16 + lr];
#pragma unroll
  for (int g = 0; g < 8; ++g) {
    int n = m0 + half * 8 + g;
    size_t base = ((size_t)b * NPIX + n) * CQKD;
    Qb[base + lr]      = f2bf(aq0[g] + bql);
    Qb[base + 16 + lr] = f2bf(aq1[g] + bqh);
    Kb[base + lr]      = f2bf(ak0[g] + bkl);
    Kb[base + 16 + lr] = f2bf(ak1[g] + bkh);
  }
}

// ---------------- kernel 3: V projection (WMMA), output channel-major ----------------
__global__ void __launch_bounds__(32) k_proj_v(
    const unsigned short* __restrict__ xp, const unsigned short* __restrict__ wv,
    const float* __restrict__ bv, unsigned short* __restrict__ Vt) {
  int l = threadIdx.x, half = l >> 4, lr = l & 15;
  int b  = blockIdx.x >> 8;
  int m0 = (blockIdx.x & 255) * 16;

  v8f zero = {};
  v8f acc[16];
#pragma unroll
  for (int i = 0; i < 16; ++i) acc[i] = zero;

  const u32x4* arow = (const u32x4*)(xp + ((size_t)(b * NPIX + m0 + lr)) * CCH);
  for (int kc = 0; kc < 8; ++kc) {
    int u = kc * 4;
    BV A;
    A.u[0] = arow[u + half * 2];  A.u[1] = arow[u + half * 2 + 1];
#pragma unroll
    for (int ct = 0; ct < 16; ++ct) {
      const u32x4* wp = (const u32x4*)(wv + (size_t)(ct * 16 + lr) * CCH);
      BV B;
      B.u[0] = wp[u + half * 2];  B.u[1] = wp[u + half * 2 + 1];
      acc[ct] = wmma_bf16(A.v, B.v, acc[ct]);
    }
  }
#pragma unroll
  for (int ct = 0; ct < 16; ++ct) {
    int c = ct * 16 + lr;
    float bvc = bv[c];
#pragma unroll
    for (int g = 0; g < 8; ++g) {
      int n = m0 + half * 8 + g;
      Vt[((size_t)b * CCH + c) * NPIX + n] = f2bf(acc[ct][g] + bvc);
    }
  }
}

// ---------------- kernel 4: streaming attention, 8 waves/block ----------------
// Block = 256 threads = 8 waves = 128 query rows. Each 32-key chunk of K (2KB)
// and V (16KB) is staged once per block into double-buffered LDS via async DMA.
__global__ void __launch_bounds__(256) k_attn(
    const unsigned short* __restrict__ Qb, const unsigned short* __restrict__ Kb,
    const unsigned short* __restrict__ Vt, const float* __restrict__ x,
    const float* __restrict__ gamma, float* __restrict__ out) {
  __shared__ __attribute__((aligned(16))) unsigned short Vl[2][CCH * 32];  // 32 KB
  __shared__ __attribute__((aligned(16))) unsigned short Kl[2][32 * 32];   //  4 KB
  __shared__ __attribute__((aligned(16))) unsigned short Pl[8][16 * 32];   //  8 KB

  int t = threadIdx.x;
  int w = t >> 5, l = t & 31;
  int half = l >> 4, lr = l & 15;
  int b  = blockIdx.x >> 5;
  int m0 = (blockIdx.x & 31) * 128 + w * 16;

  // per-chunk staging: wave w DMAs V rows c = w*32..w*32+31 (4 x b128/lane);
  // waves 0..3 additionally DMA one 16B slice of each of the 32 K rows.
  auto stage = [&](int buf, int kk) {
    int c = w * 32 + l;
    const unsigned short* gv = Vt + ((size_t)(b * CCH + c) * NPIX + kk);
    unsigned lv = lds_off(&Vl[buf][c * 32]);
#pragma unroll
    for (int j = 0; j < 4; ++j)
      async_ld_b128(lv + j * 16, (const char*)gv + j * 16);
    if (w < 4) {
      const unsigned short* gk = Kb + ((size_t)(b * NPIX + kk + l)) * CQKD;
      unsigned lk = lds_off(&Kl[buf][l * 32]);
      async_ld_b128(lk + w * 16, (const char*)gk + w * 16);
    }
  };

  BV Aq;
  {
    const u32x4* qp = (const u32x4*)(Qb + ((size_t)(b * NPIX + m0 + lr)) * CQKD);
    Aq.u[0] = qp[half * 2];  Aq.u[1] = qp[half * 2 + 1];
  }
  v8f zero = {};
  v8f acc[16];
#pragma unroll
  for (int i = 0; i < 16; ++i) acc[i] = zero;
  float rs[8];
#pragma unroll
  for (int g = 0; g < 8; ++g) rs[g] = 0.f;

  stage(0, 0);
  for (int i = 0; i < NPIX / 32; ++i) {
    int kk  = i * 32;
    int buf = i & 1;
    if (i + 1 < NPIX / 32) {
      stage(buf ^ 1, kk + 32);       // prefetch next chunk into other buffer
      if (w < 4) { WAIT_ASYNC(5); } else { WAIT_ASYNC(4); }  // current chunk done
    } else {
      WAIT_ASYNC(0);
    }
    __syncthreads();                  // all waves' DMA for `buf` visible

    // S = Q * K^T  (two 16-key subtiles) from LDS
    BV Bk0, Bk1;
    Bk0.u[0] = *(const u32x4*)&Kl[buf][lr * 32 + half * 16];
    Bk0.u[1] = *(const u32x4*)&Kl[buf][lr * 32 + half * 16 + 8];
    Bk1.u[0] = *(const u32x4*)&Kl[buf][(16 + lr) * 32 + half * 16];
    Bk1.u[1] = *(const u32x4*)&Kl[buf][(16 + lr) * 32 + half * 16 + 8];
    v8f S0 = wmma_bf16(Aq.v, Bk0.v, zero);
    v8f S1 = wmma_bf16(Aq.v, Bk1.v, zero);

    // P = exp(S) (unnormalized), row-sum partials, stage P in per-wave LDS
#pragma unroll
    for (int g = 0; g < 8; ++g) {
      float p0 = __expf(S0[g]);
      float p1 = __expf(S1[g]);
      rs[g] += p0 + p1;
      int row = half * 8 + g;
      Pl[w][row * 32 + lr]      = f2bf(p0);
      Pl[w][row * 32 + 16 + lr] = f2bf(p1);
    }
    MEMBAR();                         // D-layout -> A-layout through LDS (in-wave)
    BV Ap;
    Ap.u[0] = *(const u32x4*)&Pl[w][lr * 32 + half * 16];
    Ap.u[1] = *(const u32x4*)&Pl[w][lr * 32 + half * 16 + 8];

    // O += P * V  (16 column tiles of C=256) from LDS
#pragma unroll
    for (int ct = 0; ct < 16; ++ct) {
      BV Bv;
      int c = ct * 16 + lr;
      Bv.u[0] = *(const u32x4*)&Vl[buf][c * 32 + half * 16];
      Bv.u[1] = *(const u32x4*)&Vl[buf][c * 32 + half * 16 + 8];
      acc[ct] = wmma_bf16(Ap.v, Bv.v, acc[ct]);
    }
    __syncthreads();                  // done reading `buf` before it is re-DMAed
  }

  // all-reduce row sums across the 16 lanes of each half
#pragma unroll
  for (int g = 0; g < 8; ++g) {
    float v = rs[g];
    v += __shfl_xor(v, 1, 32);
    v += __shfl_xor(v, 2, 32);
    v += __shfl_xor(v, 4, 32);
    v += __shfl_xor(v, 8, 32);
    rs[g] = v;
  }

  float gm = gamma[0];
#pragma unroll
  for (int ct = 0; ct < 16; ++ct) {
    int c = ct * 16 + lr;
#pragma unroll
    for (int g = 0; g < 8; ++g) {
      int n = m0 + half * 8 + g;
      size_t idx = ((size_t)b * CCH + c) * NPIX + n;
      out[idx] = gm * (acc[ct][g] / rs[g]) + x[idx];
    }
  }
}

// ---------------- launch ----------------
extern "C" void kernel_launch(void* const* d_in, const int* in_sizes, int n_in,
                              void* d_out, int out_size, void* d_ws, size_t ws_size,
                              hipStream_t stream) {
  (void)in_sizes; (void)n_in; (void)out_size; (void)ws_size;
  const float* x     = (const float*)d_in[0];
  const float* Wq    = (const float*)d_in[1];
  const float* bq    = (const float*)d_in[2];
  const float* Wk    = (const float*)d_in[3];
  const float* bk    = (const float*)d_in[4];
  const float* Wv    = (const float*)d_in[5];
  const float* bv    = (const float*)d_in[6];
  const float* gamma = (const float*)d_in[7];
  float* out = (float*)d_out;

  char* ws = (char*)d_ws;
  unsigned short* wq16 = (unsigned short*)(ws + 0);         //  16 KB
  unsigned short* wk16 = (unsigned short*)(ws + 16384);     //  16 KB
  unsigned short* wv16 = (unsigned short*)(ws + 32768);     // 128 KB
  unsigned short* xp   = (unsigned short*)(ws + 163840);    //  16 MB
  unsigned short* Qb   = (unsigned short*)(ws + 16941056);  //   2 MB
  unsigned short* Kb   = (unsigned short*)(ws + 19038208);  //   2 MB
  unsigned short* Vt   = (unsigned short*)(ws + 21135360);  //  16 MB (total ~36.2 MB)

  k_wcvt<<<320, 256, 0, stream>>>(Wq, Wk, Wv, wq16, wk16, wv16);
  k_xpose<<<dim3(128, 8, 8), dim3(32, 8), 0, stream>>>(x, xp);
  k_proj_qk<<<2048, 32, 0, stream>>>(xp, wq16, wk16, bq, bk, Qb, Kb);
  k_proj_v<<<2048, 32, 0, stream>>>(xp, wv16, bv, Vt);
  k_attn<<<256, 256, 0, stream>>>(Qb, Kb, Vt, x, gamma, out);
}
